// MAFM_22445499089535
// MI455X (gfx1250) — compile-verified, matrix operand order checked by reading, not run on hardware
//
#include <hip/hip_runtime.h>
#include <math.h>

typedef __attribute__((ext_vector_type(16))) _Float16 v16h;
typedef __attribute__((ext_vector_type(8)))  float    v8f;
typedef __attribute__((ext_vector_type(4)))  float    v4f;

#define B_    32
#define H_    56
#define W_    56
#define C_    256
#define C4_   (C_/4)         // 64 float4 per channel row
#define L_    (H_*W_)        // 3136
#define S_    16             // S6 state
#define EPS_  1e-5f
#define NCH   56             // scan chunks
#define CLEN  56             // chunk length (NCH*CLEN == L_)
#define K1CH  16             // spatial-reduce chunks
#define K1P   (L_/K1CH)      // 196 positions per chunk
#define K1P4  (K1P/4)        // 49 iterations of 4 positions

// ---------------------------------------------------------------------------
// K1a: partial spatial sum/max of x per (b, chunk, sub, c4). 256 thr =
// 4 position-subgroups x 64 float4-channel-owners -> b128 loads.
// ---------------------------------------------------------------------------
__global__ __launch_bounds__(256) void k1a_spatial_partial(
    const float4* __restrict__ x4, float4* __restrict__ sp4, float4* __restrict__ mp4)
{
    const int b = blockIdx.x, pc = blockIdx.y;
    const int sub = threadIdx.x >> 6;        // 0..3 position subgroup
    const int c4  = threadIdx.x & 63;        // float4 channel group
    float4 s = make_float4(0.f, 0.f, 0.f, 0.f);
    float4 m = make_float4(-INFINITY, -INFINITY, -INFINITY, -INFINITY);
    const int p0 = pc*K1P;
    for (int i = 0; i < K1P4; ++i) {
        const int pp = b*L_ + p0 + i*4 + sub;
        const float4 v = x4[(size_t)pp*C4_ + c4];
        s.x += v.x; s.y += v.y; s.z += v.z; s.w += v.w;
        m.x = fmaxf(m.x, v.x); m.y = fmaxf(m.y, v.y);
        m.z = fmaxf(m.z, v.z); m.w = fmaxf(m.w, v.w);
    }
    const int row = (b*K1CH + pc)*4 + sub;   // 64 rows per batch
    sp4[row*C4_ + c4] = s;
    mp4[row*C4_ + c4] = m;
}

// K1b: finish -> avg[b][c], mx[b][c]
__global__ __launch_bounds__(256) void k1b_spatial_final(
    const float* __restrict__ sp, const float* __restrict__ mp,
    float* __restrict__ avg, float* __restrict__ mx)
{
    const int b = blockIdx.x, c = threadIdx.x;
    float s = 0.f, m = -INFINITY;
    for (int r = 0; r < K1CH*4; ++r) {
        s += sp[(b*K1CH*4 + r)*C_ + c];
        m = fmaxf(m, mp[(b*K1CH*4 + r)*C_ + c]);
    }
    avg[b*C_ + c] = s * (1.f / (float)L_);
    mx[b*C_ + c]  = m;
}

// ---------------------------------------------------------------------------
// K2: channel-attention MLP via WMMA.
//   G1: [avg;mx] (64x256) @ W1^T (256x16) -> relu -> fold -> hsum (32x16)
//   G2: hsum (32x16) @ W2^T (16x256) -> sigmoid -> gate (32x256)
// One block, 4 waves. Lane layouts follow CDNA5 ISA 7.12.2.
// ---------------------------------------------------------------------------
__device__ __forceinline__ int wmma_k(int e, int half_) {
    return (e < 8) ? (half_*8 + e) : (16 + half_*8 + (e - 8));
}

__global__ __launch_bounds__(128) void k2_ca_fc_wmma(
    const float* __restrict__ avg, const float* __restrict__ mx,
    const float* __restrict__ w1,  const float* __restrict__ w2,
    float* __restrict__ gate)
{
    __shared__ float    h1[64*16];
    __shared__ _Float16 hs[32*16];
    const int tid  = threadIdx.x;
    const int wave = tid >> 5;
    const int lane = tid & 31;
    const int half_= lane >> 4;
    const int l16  = lane & 15;

    // ---- GEMM1 : one 16-row tile per wave, K loop 256/32 (unrolled) ----
    v8f acc = {};
    const int R = wave*16 + l16;                       // stacked row (0..63)
    const float* srcRow = (R < 32) ? (avg + R*C_) : (mx + (R-32)*C_);
    #pragma unroll
    for (int kk = 0; kk < 8; ++kk) {
        const int k0 = kk*32;
        v16h a, b;
        #pragma unroll
        for (int e = 0; e < 16; ++e) {
            const int k = k0 + wmma_k(e, half_);
            a[e] = (_Float16)srcRow[k];
            b[e] = (_Float16)w1[l16*C_ + k];           // B[k][n] = W1[n][k]
        }
        acc = __builtin_amdgcn_wmma_f32_16x16x32_f16(
                false, a, false, b, (short)0, acc, false, false);
    }
    #pragma unroll
    for (int r = 0; r < 8; ++r) {                      // D: row = r + half*8
        const int grow = wave*16 + r + half_*8;
        h1[grow*16 + l16] = fmaxf(acc[r], 0.f);
    }
    __syncthreads();
    for (int i = tid; i < 512; i += 128)               // relu(avg)+relu(mx)
        hs[i] = (_Float16)(h1[i] + h1[i + 512]);
    __syncthreads();

    // ---- GEMM2 : 2 M-tiles x 16 N-tiles, 8 tiles per wave, K=16 (padded) ----
    for (int it = 0; it < 8; ++it) {
        const int tileid = wave*8 + it;
        const int mt = tileid >> 4;                    // 0..1
        const int nt = tileid & 15;                    // 0..15
        const int n  = nt*16 + l16;                    // output channel
        v16h a, b;
        #pragma unroll
        for (int e = 0; e < 16; ++e) {
            const int k = wmma_k(e, half_);
            a[e] = (k < 16) ? hs[(mt*16 + l16)*16 + k] : (_Float16)0.f;
            b[e] = (k < 16) ? (_Float16)w2[n*16 + k]   : (_Float16)0.f;
        }
        v8f c = {};
        c = __builtin_amdgcn_wmma_f32_16x16x32_f16(
                false, a, false, b, (short)0, c, false, false);
        #pragma unroll
        for (int r = 0; r < 8; ++r) {
            const int batch = mt*16 + r + half_*8;
            gate[batch*C_ + n] = 1.f / (1.f + __expf(-c[r]));
        }
    }
}

// ---------------------------------------------------------------------------
// K3: per-pixel channel mean/max of xc = x*gate. One wave per pixel; each
// lane owns 8 contiguous channels (2 x b128 loads).
// ---------------------------------------------------------------------------
__global__ __launch_bounds__(256) void k3_pixel_reduce(
    const float4* __restrict__ x4, const float4* __restrict__ gate4,
    float* __restrict__ m, float* __restrict__ cmax)
{
    const int wave = threadIdx.x >> 5, lane = threadIdx.x & 31;
    const int pp = blockIdx.x*8 + wave;                // global pixel (b*L+p)
    const int b  = pp / L_;
    float s = 0.f, mm = -INFINITY;
    #pragma unroll
    for (int i = 0; i < 2; ++i) {
        const float4 xv = x4[(size_t)pp*C4_ + lane*2 + i];
        const float4 gv = gate4[b*C4_ + lane*2 + i];
        const float v0 = xv.x*gv.x, v1 = xv.y*gv.y, v2 = xv.z*gv.z, v3 = xv.w*gv.w;
        s += v0 + v1 + v2 + v3;
        mm = fmaxf(mm, fmaxf(fmaxf(v0, v1), fmaxf(v2, v3)));
    }
    #pragma unroll
    for (int d = 16; d > 0; d >>= 1) {
        s  += __shfl_xor(s, d, 32);
        mm  = fmaxf(mm, __shfl_xor(mm, d, 32));
    }
    if (lane == 0) { m[pp] = s * (1.f/(float)C_); cmax[pp] = mm; }
}

// ---------------------------------------------------------------------------
// K4a: precompute scan coefficients a,b,c per (b,p,s)  (fully parallel)
//   a = exp(delta*A[s]), b = delta*m*(m*Bw[s]+Bb[s]), c = m*Cw[s]+Cb[s]
// ---------------------------------------------------------------------------
__global__ __launch_bounds__(256) void k4a_coeffs(
    const float* __restrict__ m, const float* __restrict__ A,
    const float* __restrict__ dw, const float* __restrict__ db,
    const float* __restrict__ Bw, const float* __restrict__ Bb,
    const float* __restrict__ Cw, const float* __restrict__ Cb,
    float4* __restrict__ abc)
{
    const int gid = blockIdx.x*256 + threadIdx.x;      // (b*L+p)*16 + s
    const int s  = gid & 15;
    const int bp = gid >> 4;
    const float mv = m[bp];
    const float z  = mv * dw[0] + db[0];
    const float delta = (z > 20.f) ? z : log1pf(__expf(z));
    const float a  = __expf(delta * A[s]);
    const float bm = mv * Bw[s] + Bb[s];
    const float bq = delta * mv * bm;
    const float cc = mv * Cw[s] + Cb[s];
    abc[gid] = make_float4(a, bq, cc, 0.f);
}

// K4b1: per-chunk summaries (Aprod, Bcomb) for h' = a*h + b
__global__ void k4b1_chunk_summary(
    const float4* __restrict__ abc, float* __restrict__ Ach, float* __restrict__ Bch)
{
    const int s = threadIdx.x;                          // 0..15
    const int ch = blockIdx.y*16 + threadIdx.y;
    const int b = blockIdx.x;
    if (ch >= NCH) return;
    float h = 0.f, Ap = 1.f;
    for (int i = 0; i < CLEN; ++i) {
        const float4 f = abc[((size_t)(b*L_ + ch*CLEN + i))*S_ + s];
        h = f.x*h + f.y;
        Ap *= f.x;
    }
    Ach[(b*S_ + s)*NCH + ch] = Ap;
    Bch[(b*S_ + s)*NCH + ch] = h;
}

// K4b2: sequential scan over 56 chunk summaries -> per-chunk h_in
__global__ __launch_bounds__(512) void k4b2_chunk_scan(
    const float* __restrict__ Ach, const float* __restrict__ Bch,
    float* __restrict__ Hin)
{
    const int t = threadIdx.x;                          // (b,s)
    const int base = t*NCH;
    float h = 0.f;
    for (int ch = 0; ch < NCH; ++ch) {
        Hin[base + ch] = h;
        h = Ach[base + ch]*h + Bch[base + ch];
    }
}

// K4b3: replay chunks with correct h_in, reduce over states -> y[b][p]
__global__ void k4b3_emit_y(
    const float4* __restrict__ abc, const float* __restrict__ Hin,
    float* __restrict__ y)
{
    const int s = threadIdx.x;                          // 0..15 (lane group)
    const int ch = blockIdx.y*8 + threadIdx.y;          // 0..55
    const int b = blockIdx.x;
    float h = Hin[(b*S_ + s)*NCH + ch];
    for (int i = 0; i < CLEN; ++i) {
        const int p = ch*CLEN + i;
        const float4 f = abc[((size_t)(b*L_ + p))*S_ + s];
        h = f.x*h + f.y;
        float t = h * f.z;
        t += __shfl_xor(t, 1, 16);
        t += __shfl_xor(t, 2, 16);
        t += __shfl_xor(t, 4, 16);
        t += __shfl_xor(t, 8, 16);
        if (s == 0) y[b*L_ + p] = t;
    }
}

// ---------------------------------------------------------------------------
// K5: 7x7 SAME conv on (m, cmax) + fuse both sigmoids: gf = sig(y)+sig(s)
// ---------------------------------------------------------------------------
__global__ __launch_bounds__(256) void k5_lsa_conv(
    const float* __restrict__ m, const float* __restrict__ cmax,
    const float* __restrict__ lsa, const float* __restrict__ y,
    float* __restrict__ gf)
{
    const int pp = blockIdx.x*256 + threadIdx.x;        // b*L + p
    const int b = pp / L_, p = pp % L_;
    const int h = p / W_, w = p % W_;
    float acc = 0.f;
    #pragma unroll
    for (int kh = 0; kh < 7; ++kh) {
        const int hh = h + kh - 3;
        if (hh < 0 || hh >= H_) continue;
        #pragma unroll
        for (int kw = 0; kw < 7; ++kw) {
            const int ww = w + kw - 3;
            if (ww < 0 || ww >= W_) continue;
            const int idx = b*L_ + hh*W_ + ww;
            acc += m[idx]   * lsa[(kh*7 + kw)*2 + 0]
                 + cmax[idx]* lsa[(kh*7 + kw)*2 + 1];
        }
    }
    const float sy = 1.f / (1.f + __expf(-y[pp]));
    const float ss = 1.f / (1.f + __expf(-acc));
    gf[pp] = sy + ss;
}

// ---------------------------------------------------------------------------
// K6a: per-channel partial sum / sumsq of x2 = x*gate*gf (b128 loads)
// ---------------------------------------------------------------------------
__global__ __launch_bounds__(256) void k6a_stats_partial(
    const float4* __restrict__ x4, const float4* __restrict__ gate4,
    const float* __restrict__ gf, float4* __restrict__ psum4, float4* __restrict__ psq4)
{
    const int b = blockIdx.x, pc = blockIdx.y;
    const int sub = threadIdx.x >> 6;        // 0..3
    const int c4  = threadIdx.x & 63;
    const float4 g = gate4[b*C4_ + c4];
    float4 s = make_float4(0.f, 0.f, 0.f, 0.f);
    float4 q = make_float4(0.f, 0.f, 0.f, 0.f);
    const int p0 = pc*K1P;
    for (int i = 0; i < K1P4; ++i) {
        const int pp = b*L_ + p0 + i*4 + sub;
        const float4 xv = x4[(size_t)pp*C4_ + c4];
        const float gv = gf[pp];
        const float v0 = xv.x*g.x*gv, v1 = xv.y*g.y*gv;
        const float v2 = xv.z*g.z*gv, v3 = xv.w*g.w*gv;
        s.x += v0; s.y += v1; s.z += v2; s.w += v3;
        q.x += v0*v0; q.y += v1*v1; q.z += v2*v2; q.w += v3*v3;
    }
    const int row = (b*K1CH + pc)*4 + sub;
    psum4[row*C4_ + c4] = s;
    psq4 [row*C4_ + c4] = q;
}

// K6b: fold partials -> BN scale/bias
__global__ __launch_bounds__(256) void k6b_stats_final(
    const float* __restrict__ psum, const float* __restrict__ psq,
    const float* __restrict__ gamma, const float* __restrict__ beta,
    float* __restrict__ scale, float* __restrict__ bias)
{
    const int c = threadIdx.x;
    float s = 0.f, q = 0.f;
    for (int i = 0; i < B_*K1CH*4; ++i) { s += psum[i*C_ + c]; q += psq[i*C_ + c]; }
    const float inv_n = 1.f / (float)(B_*L_);
    const float mean = s * inv_n;
    const float var  = q * inv_n - mean*mean;
    const float sc = gamma[c] * rsqrtf(var + EPS_);
    scale[c] = sc;
    bias[c]  = beta[c] - mean * sc;
}

// ---------------------------------------------------------------------------
// K7: out = relu( x*gate*gf*scale[c] + bias[c] )   (b128 loads, NT stores)
// ---------------------------------------------------------------------------
__global__ __launch_bounds__(256) void k7_final(
    const float4* __restrict__ x4, const float4* __restrict__ gate4,
    const float* __restrict__ gf, const float4* __restrict__ scale4,
    const float4* __restrict__ bias4, float4* __restrict__ out4)
{
    const int pp = blockIdx.x*4 + (threadIdx.x >> 6);   // b*L + p
    const int c4 = threadIdx.x & 63;
    const int b  = pp / L_;
    const float4 xv = x4[(size_t)pp*C4_ + c4];
    const float4 g  = gate4[b*C4_ + c4];
    const float4 sc = scale4[c4];
    const float4 bs = bias4[c4];
    const float gv  = gf[pp];
    v4f o;
    o.x = fmaxf(xv.x*g.x*gv*sc.x + bs.x, 0.f);
    o.y = fmaxf(xv.y*g.y*gv*sc.y + bs.y, 0.f);
    o.z = fmaxf(xv.z*g.z*gv*sc.z + bs.z, 0.f);
    o.w = fmaxf(xv.w*g.w*gv*sc.w + bs.w, 0.f);
    __builtin_nontemporal_store(o, (v4f*)&out4[(size_t)pp*C4_ + c4]);
}

// ---------------------------------------------------------------------------
extern "C" void kernel_launch(void* const* d_in, const int* in_sizes, int n_in,
                              void* d_out, int out_size, void* d_ws, size_t ws_size,
                              hipStream_t stream) {
    const float* x     = (const float*)d_in[0];
    const float* ca_w1 = (const float*)d_in[1];
    const float* ca_w2 = (const float*)d_in[2];
    const float* lsa_w = (const float*)d_in[3];
    const float* A     = (const float*)d_in[4];
    const float* d_w   = (const float*)d_in[5];
    const float* d_b   = (const float*)d_in[6];
    const float* B_w   = (const float*)d_in[7];
    const float* B_b   = (const float*)d_in[8];
    const float* C_w   = (const float*)d_in[9];
    const float* C_b   = (const float*)d_in[10];
    const float* gamma = (const float*)d_in[11];
    const float* beta  = (const float*)d_in[12];
    float* out = (float*)d_out;

    // workspace carving (floats); all region sizes are multiples of 4 floats
    float* ws = (float*)d_ws;
    float* avg   = ws;                 // 32*256
    float* mx    = avg   + B_*C_;
    float* gate  = mx    + B_*C_;
    float* sp    = gate  + B_*C_;      // 32*16*4*256
    float* mp    = sp    + B_*K1CH*4*C_;
    float* m     = mp    + B_*K1CH*4*C_; // 32*3136
    float* cmax  = m     + B_*L_;
    float* y     = cmax  + B_*L_;
    float* gf    = y     + B_*L_;
    float* abc_f = gf    + B_*L_;      // float4 x (32*3136*16)
    float4* abc  = (float4*)abc_f;
    float* Ach   = abc_f + (size_t)B_*L_*S_*4;  // 32*16*56
    float* Bch   = Ach   + B_*S_*NCH;
    float* Hin   = Bch   + B_*S_*NCH;
    float* psum  = Hin   + B_*S_*NCH;  // 32*16*4*256
    float* psq   = psum  + B_*K1CH*4*C_;
    float* scale = psq   + B_*K1CH*4*C_;
    float* bias  = scale + C_;

    // 1) spatial avg/max per (b,c)
    k1a_spatial_partial<<<dim3(B_, K1CH), 256, 0, stream>>>(
        (const float4*)x, (float4*)sp, (float4*)mp);
    k1b_spatial_final<<<B_, 256, 0, stream>>>(sp, mp, avg, mx);

    // 2) channel-attention MLP via WMMA -> gate
    k2_ca_fc_wmma<<<1, 128, 0, stream>>>(avg, mx, ca_w1, ca_w2, gate);

    // 3) per-pixel channel mean/max of xc
    k3_pixel_reduce<<<(B_*L_)/8, 256, 0, stream>>>(
        (const float4*)x, (const float4*)gate, m, cmax);

    // 4) S6 scan (parallel chunked)
    k4a_coeffs<<<(B_*L_*S_)/256, 256, 0, stream>>>(m, A, d_w, d_b, B_w, B_b, C_w, C_b, abc);
    k4b1_chunk_summary<<<dim3(B_, 4), dim3(16, 16), 0, stream>>>(abc, Ach, Bch);
    k4b2_chunk_scan<<<1, B_*S_, 0, stream>>>(Ach, Bch, Hin);
    k4b3_emit_y<<<dim3(B_, 7), dim3(16, 8), 0, stream>>>(abc, Hin, y);

    // 5) LSA conv + fused sigmoids -> gf = sig(y)+sig(s)
    k5_lsa_conv<<<(B_*L_)/256, 256, 0, stream>>>(m, cmax, lsa_w, y, gf);

    // 6) BN stats over x2 (recomputed from scalars)
    k6a_stats_partial<<<dim3(B_, K1CH), 256, 0, stream>>>(
        (const float4*)x, (const float4*)gate, gf, (float4*)psum, (float4*)psq);
    k6b_stats_final<<<1, 256, 0, stream>>>(psum, psq, gamma, beta, scale, bias);

    // 7) final fused normalize + relu
    k7_final<<<(B_*L_)/4, 256, 0, stream>>>(
        (const float4*)x, (const float4*)gate, gf,
        (const float4*)scale, (const float4*)bias, (float4*)out);
}
